// MusicVAE_29678224015655
// MI455X (gfx1250) — compile-verified, matrix operand order checked by reading, not run on hardware
//
#include <hip/hip_runtime.h>
#include <hip/hip_bf16.h>
#include <math.h>

// ---------------------------------------------------------------------------
// MusicVAE forward for MI455X (gfx1250, wave32, WMMA).
// All large GEMMs run on v_wmma_f32_16x16x32_bf16 (bf16 A/B, f32 accum).
// One-hot input matmuls are implemented as weight-column gathers.
// GEMM inner loop is double-buffered (load k+1 while computing k) with a
// predication-free fast path for fully in-bounds N tiles.
// ---------------------------------------------------------------------------

typedef __bf16 bf16_t;
typedef __attribute__((ext_vector_type(16))) __bf16 v16bf;
typedef __attribute__((ext_vector_type(8)))  float  v8f;

#define DV   258
#define DT   32
#define DB   512
#define DHE  2048
#define DZ   512
#define DHC  1024
#define DHD  1024
#define DU   2
#define DSUB 16

union FragBf { v16bf v; unsigned int u[8]; uint4 q[2]; };

__device__ __forceinline__ float sigm(float x) { return 1.0f / (1.0f + __expf(-x)); }
__device__ __forceinline__ float softplusf(float x) {
  return (x > 0.f) ? (x + log1pf(__expf(-x))) : log1pf(__expf(x));
}

// ---------------------------------------------------------------------------
// WMMA GEMM:  C[M,N] = base + A[M,K](bf16,row-major,lda) * W[N,K](bf16,ldw)^T
// mode 0: base = bias[n] (bias may be null);  mode 1: base = Cin[m,n]
// M % 64 == 0, K % 32 == 0, lda/ldw multiples of 8 elements (16B-aligned rows).
// Block tile 64x256, 8 waves, wave tile 32x64 (2x4 WMMA tiles).
// Fragment layouts per CDNA5 ISA 7.12.2:
//   16-bit A: lanes 0-15 row M=lane hold K {0..7,16..23}; lanes 16-31 hold
//   K {8..15,24..31}. B symmetric with N. C/D: VGPR r -> M = r (+8 for
//   lanes>=16), N = lane&15.
// ---------------------------------------------------------------------------

struct GemmPtrs { const bf16_t* a[2]; const bf16_t* b[4]; };

__device__ __forceinline__ void load_frags(FragBf (&af)[2], FragBf (&wf)[4],
                                           const GemmPtrs& p)
{
#pragma unroll
  for (int mt = 0; mt < 2; ++mt) {
    af[mt].q[0] = *(const uint4*)(p.a[mt]);
    af[mt].q[1] = *(const uint4*)(p.a[mt] + 16);
  }
#pragma unroll
  for (int nt = 0; nt < 4; ++nt) {
    wf[nt].q[0] = *(const uint4*)(p.b[nt]);
    wf[nt].q[1] = *(const uint4*)(p.b[nt] + 16);
  }
}

__device__ __forceinline__ void advance(GemmPtrs& p)
{
#pragma unroll
  for (int mt = 0; mt < 2; ++mt) p.a[mt] += 32;
#pragma unroll
  for (int nt = 0; nt < 4; ++nt) p.b[nt] += 32;
}

__device__ __forceinline__ void mma_all(v8f (&acc)[2][4], const FragBf (&af)[2],
                                        const FragBf (&wf)[4])
{
#pragma unroll
  for (int mt = 0; mt < 2; ++mt)
#pragma unroll
    for (int nt = 0; nt < 4; ++nt)
      acc[mt][nt] = __builtin_amdgcn_wmma_f32_16x16x32_bf16(
          false, af[mt].v, false, wf[nt].v, (short)0, acc[mt][nt], false, false);
}

__global__ __launch_bounds__(256) void gemm_bf16_k(
    const bf16_t* __restrict__ A, int lda,
    const bf16_t* __restrict__ W, int ldw,
    const float* __restrict__ Cin,
    const float* __restrict__ bias,
    float* __restrict__ C, int ldc,
    int N, int K, int mode)
{
  const int lane    = threadIdx.x & 31;
  const int wave    = threadIdx.x >> 5;
  const int wm      = wave >> 2;               // 0..1
  const int wn      = wave & 3;                // 0..3
  const int rowBase = blockIdx.y * 64 + wm * 32;
  const int colBase = blockIdx.x * 256 + wn * 64;
  const int l15     = lane & 15;
  const int khalf   = (lane >> 4) << 3;        // 0 or 8

  v8f acc[2][4];
#pragma unroll
  for (int mt = 0; mt < 2; ++mt)
#pragma unroll
    for (int nt = 0; nt < 4; ++nt)
#pragma unroll
      for (int r = 0; r < 8; ++r) acc[mt][nt][r] = 0.f;

  if (colBase + 64 <= N) {
    // ---------------- fast path: no predication, double-buffered -----------
    GemmPtrs p;
#pragma unroll
    for (int mt = 0; mt < 2; ++mt)
      p.a[mt] = A + (size_t)(rowBase + mt * 16 + l15) * lda + khalf;
#pragma unroll
    for (int nt = 0; nt < 4; ++nt)
      p.b[nt] = W + (size_t)(colBase + nt * 16 + l15) * ldw + khalf;

    FragBf a0[2], b0[4], a1[2], b1[4];
    load_frags(a0, b0, p);
    advance(p);
    int rem = (K >> 5) - 1;                 // k-tiles remaining after first
    while (rem >= 2) {
      load_frags(a1, b1, p); advance(p);
      mma_all(acc, a0, b0);
      load_frags(a0, b0, p); advance(p);
      mma_all(acc, a1, b1);
      rem -= 2;
    }
    if (rem == 1) {
      load_frags(a1, b1, p);
      mma_all(acc, a0, b0);
      mma_all(acc, a1, b1);
    } else {
      mma_all(acc, a0, b0);
    }
  } else {
    // ---------------- edge path (only the N=258 head GEMM) -----------------
    for (int k0 = 0; k0 < K; k0 += 32) {
      FragBf af[2], wf[4];
#pragma unroll
      for (int mt = 0; mt < 2; ++mt) {
        const bf16_t* ap = A + (size_t)(rowBase + mt * 16 + l15) * lda + (k0 + khalf);
        af[mt].q[0] = *(const uint4*)(ap);
        af[mt].q[1] = *(const uint4*)(ap + 16);
      }
#pragma unroll
      for (int nt = 0; nt < 4; ++nt) {
        int n = colBase + nt * 16 + l15;
        if (n < N) {
          const bf16_t* bp = W + (size_t)n * ldw + (k0 + khalf);
          wf[nt].q[0] = *(const uint4*)(bp);
          wf[nt].q[1] = *(const uint4*)(bp + 16);
        } else {
#pragma unroll
          for (int i = 0; i < 8; ++i) wf[nt].u[i] = 0u;
        }
      }
      mma_all(acc, af, wf);
    }
  }

  // ---------------- epilogue ----------------
  const int mh = (lane >> 4) << 3;             // 0 or 8
#pragma unroll
  for (int mt = 0; mt < 2; ++mt) {
#pragma unroll
    for (int nt = 0; nt < 4; ++nt) {
      int n = colBase + nt * 16 + l15;
      if (n >= N) continue;
#pragma unroll
      for (int r = 0; r < 8; ++r) {
        int m = rowBase + mt * 16 + mh + r;
        float add = (mode == 0) ? (bias ? bias[n] : 0.f)
                                : Cin[(size_t)m * ldc + n];
        C[(size_t)m * ldc + n] = acc[mt][nt][r] + add;
      }
    }
  }
}

// fp32 -> bf16 with row restride + zero-padding of K up to Kpad.
__global__ void cvt_pad_k(const float* __restrict__ src, bf16_t* __restrict__ dst,
                          int rows, int K, int srcStride, int Kpad)
{
  long i = (long)blockIdx.x * blockDim.x + threadIdx.x;
  long total = (long)rows * Kpad;
  if (i >= total) return;
  int r = (int)(i / Kpad), k = (int)(i % Kpad);
  float v = (k < K) ? src[(size_t)r * srcStride + k] : 0.f;
  dst[i] = (bf16_t)v;
}

// Encoder one-hot input: gates[b,g] = bias[g] + Wih[g, x[b,col]]
__global__ void enc_gate_init_k(float* __restrict__ gates,
                                const float* __restrict__ Wih, int ldw,
                                const float* __restrict__ bias,
                                const int* __restrict__ x, int T, int col, int G)
{
  int i = blockIdx.x * blockDim.x + threadIdx.x;  // over M*G (<= 4.2M)
  int m = i / G, g = i - m * G;
  int tok = x[m * T + col];
  gates[(size_t)i] = bias[g] + Wih[(size_t)g * ldw + tok];
}

// Decoder: gates[(u,b),g] = cpart[(u,b),g] + Wih[g, 512 + token(u,b,s)]
__global__ void dec_gate_init_k(float* __restrict__ gates,
                                const float* __restrict__ cpart,
                                const float* __restrict__ Wih, int ldw, int colOff,
                                const int* __restrict__ x, int T, int s, int G)
{
  int i = blockIdx.x * blockDim.x + threadIdx.x;  // over 1024*G
  int m = i / G, g = i - m * G;
  int u = m >> 9, b = m & 511;
  int t = u * DSUB + s;
  int tok = (t == 0) ? 0 : x[b * T + (t - 1)];
  gates[(size_t)i] = cpart[(size_t)i] + Wih[(size_t)g * ldw + colOff + tok];
}

// LSTM cell: gate order i,f,g,o (torch). c updated in place (f32), h written
// as bf16 (next-step GEMM input) and optionally to a secondary bf16 location:
//   sec[ ((m/secMod)*secBlk + m%secMod) * secRow + j ]
__global__ void lstm_cell_k(const float* __restrict__ gates, float* __restrict__ c,
                            bf16_t* __restrict__ h,
                            bf16_t* __restrict__ sec, int secMod, long secBlk,
                            long secRow, int M, int H)
{
  int i = blockIdx.x * blockDim.x + threadIdx.x;
  if (i >= M * H) return;
  int m = i / H, j = i - m * H;
  const float* gm = gates + (size_t)m * (4 * H);
  float ig = sigm(gm[j]);
  float fg = sigm(gm[H + j]);
  float gg = tanhf(gm[2 * H + j]);
  float og = sigm(gm[3 * H + j]);
  float cn = fg * c[i] + ig * gg;
  c[i] = cn;
  float hv = og * tanhf(cn);
  bf16_t hb = (bf16_t)hv;
  h[i] = hb;
  if (sec) {
    long rr = (long)(m / secMod) * secBlk + (m % secMod);
    sec[rr * secRow + j] = hb;
  }
}

__global__ void tanh_bf16_k(const float* __restrict__ src, bf16_t* __restrict__ dst,
                            long n)
{
  long i = (long)blockIdx.x * blockDim.x + threadIdx.x;
  if (i >= n) return;
  dst[i] = (bf16_t)tanhf(src[i]);
}

// VAE head: mu/sigma outputs + z (bf16) for conductor init.
__global__ void vae_head_k(const float* __restrict__ muRaw,
                           const float* __restrict__ sigRaw,
                           const float* __restrict__ eps,
                           float* __restrict__ outMu, float* __restrict__ outSig,
                           bf16_t* __restrict__ zb, long n)
{
  long i = (long)blockIdx.x * blockDim.x + threadIdx.x;
  if (i >= n) return;
  float m = muRaw[i];
  float s = softplusf(sigRaw[i]);
  outMu[i]  = m;
  outSig[i] = s;
  zb[i] = (bf16_t)(m + s * eps[i]);
}

// log_softmax over V=258, with [t*B+b] -> out[b,t,:] transpose.
__global__ void logsoftmax_k(const float* __restrict__ logits,
                             float* __restrict__ out, int T, int B, int V)
{
  int r = blockIdx.x;                // r = t*B + b
  int t = r / B, b = r - t * B;
  const float* row = logits + (size_t)r * V;
  __shared__ float red[128];
  float mx = -INFINITY;
  for (int v = threadIdx.x; v < V; v += 128) mx = fmaxf(mx, row[v]);
  red[threadIdx.x] = mx; __syncthreads();
  for (int s = 64; s > 0; s >>= 1) {
    if (threadIdx.x < s) red[threadIdx.x] = fmaxf(red[threadIdx.x], red[threadIdx.x + s]);
    __syncthreads();
  }
  mx = red[0]; __syncthreads();
  float sm = 0.f;
  for (int v = threadIdx.x; v < V; v += 128) sm += __expf(row[v] - mx);
  red[threadIdx.x] = sm; __syncthreads();
  for (int s = 64; s > 0; s >>= 1) {
    if (threadIdx.x < s) red[threadIdx.x] += red[threadIdx.x + s];
    __syncthreads();
  }
  float lse = logf(red[0]) + mx;
  float* o = out + (size_t)b * T * V + (size_t)t * V;
  for (int v = threadIdx.x; v < V; v += 128) o[v] = row[v] - lse;
}

// ---------------------------------------------------------------------------
extern "C" void kernel_launch(void* const* d_in, const int* in_sizes, int n_in,
                              void* d_out, int out_size, void* d_ws, size_t ws_size,
                              hipStream_t stream)
{
  (void)in_sizes; (void)n_in; (void)out_size; (void)ws_size;

  const int*   x        = (const int*)d_in[0];
  const float* eps      = (const float*)d_in[1];
  const float* e0f_Wih  = (const float*)d_in[2];
  const float* e0f_Whh  = (const float*)d_in[3];
  const float* e0f_b    = (const float*)d_in[4];
  const float* e0b_Wih  = (const float*)d_in[5];
  const float* e0b_Whh  = (const float*)d_in[6];
  const float* e0b_b    = (const float*)d_in[7];
  const float* e1f_Wih  = (const float*)d_in[8];
  const float* e1f_Whh  = (const float*)d_in[9];
  const float* e1f_b    = (const float*)d_in[10];
  const float* e1b_Wih  = (const float*)d_in[11];
  const float* e1b_Whh  = (const float*)d_in[12];
  const float* e1b_b    = (const float*)d_in[13];
  const float* mu_W     = (const float*)d_in[14];
  const float* mu_b     = (const float*)d_in[15];
  const float* sig_W    = (const float*)d_in[16];
  const float* sig_b    = (const float*)d_in[17];
  const float* ci_W     = (const float*)d_in[18];
  const float* ci_b     = (const float*)d_in[19];
  const float* cond     = (const float*)d_in[20];
  const float* c0_Wih   = (const float*)d_in[21];
  const float* c0_Whh   = (const float*)d_in[22];
  const float* c0_b     = (const float*)d_in[23];
  const float* c1_Wih   = (const float*)d_in[24];
  const float* c1_Whh   = (const float*)d_in[25];
  const float* c1_b     = (const float*)d_in[26];
  const float* co_W     = (const float*)d_in[27];
  const float* co_b     = (const float*)d_in[28];
  const float* di_W     = (const float*)d_in[29];
  const float* di_b     = (const float*)d_in[30];
  const float* d0_Wih   = (const float*)d_in[31];
  const float* d0_Whh   = (const float*)d_in[32];
  const float* d0_b     = (const float*)d_in[33];
  const float* d1_Wih   = (const float*)d_in[34];
  const float* d1_Whh   = (const float*)d_in[35];
  const float* d1_b     = (const float*)d_in[36];
  const float* head_W   = (const float*)d_in[37];
  const float* head_b   = (const float*)d_in[38];

  float* out    = (float*)d_out;
  float* outMu  = out + (size_t)DB * DT * DV;          // probs = B*T*V
  float* outSig = outMu + (size_t)DB * DZ;

  // ---- workspace bump allocator (~600 MB total) ----
  char*  base = (char*)d_ws;
  size_t off  = 0;
  auto alloc = [&](size_t bytes) -> void* {
    off = (off + 255) & ~(size_t)255;
    void* p = base + off;
    off += bytes;
    return p;
  };

  // bf16 weight copies (zero-padded K where needed)
  bf16_t* wE0fWhh = (bf16_t*)alloc((size_t)8192 * 2048 * 2);
  bf16_t* wE0bWhh = (bf16_t*)alloc((size_t)8192 * 2048 * 2);
  bf16_t* wE1fWih = (bf16_t*)alloc((size_t)8192 * 4096 * 2);
  bf16_t* wE1fWhh = (bf16_t*)alloc((size_t)8192 * 2048 * 2);
  bf16_t* wE1bWih = (bf16_t*)alloc((size_t)8192 * 4096 * 2);
  bf16_t* wE1bWhh = (bf16_t*)alloc((size_t)8192 * 2048 * 2);
  bf16_t* wMu     = (bf16_t*)alloc((size_t)512 * 4096 * 2);
  bf16_t* wSig    = (bf16_t*)alloc((size_t)512 * 4096 * 2);
  bf16_t* wCi     = (bf16_t*)alloc((size_t)1024 * 512 * 2);
  bf16_t* wC0Wih  = (bf16_t*)alloc((size_t)4096 * 288 * 2);   // K 258 -> 288
  bf16_t* wC0Whh  = (bf16_t*)alloc((size_t)4096 * 1024 * 2);
  bf16_t* wC1Wih  = (bf16_t*)alloc((size_t)4096 * 1024 * 2);
  bf16_t* wC1Whh  = (bf16_t*)alloc((size_t)4096 * 1024 * 2);
  bf16_t* wCo     = (bf16_t*)alloc((size_t)512 * 1024 * 2);
  bf16_t* wDi     = (bf16_t*)alloc((size_t)1024 * 512 * 2);
  bf16_t* wD0Wc   = (bf16_t*)alloc((size_t)4096 * 512 * 2);   // cols 0..511 of d0_Wih
  bf16_t* wD0Whh  = (bf16_t*)alloc((size_t)4096 * 1024 * 2);
  bf16_t* wD1Wih  = (bf16_t*)alloc((size_t)4096 * 1024 * 2);
  bf16_t* wD1Whh  = (bf16_t*)alloc((size_t)4096 * 1024 * 2);
  bf16_t* wHead   = (bf16_t*)alloc((size_t)258 * 1024 * 2);
  bf16_t* condB   = (bf16_t*)alloc((size_t)512 * 288 * 2);    // cond_input padded

  // activations / state
  bf16_t* y0     = (bf16_t*)alloc((size_t)DT * DB * (2 * DHE) * 2);  // 128 MB
  float*  gates  = (float*) alloc((size_t)1024 * 4096 * 4);          // 16 MB (reused)
  bf16_t* h0s    = (bf16_t*)alloc((size_t)1024 * 2048 * 2);          // layer-0 h
  bf16_t* h1s    = (bf16_t*)alloc((size_t)1024 * 2048 * 2);          // layer-1 h
  float*  c0s    = (float*) alloc((size_t)1024 * 2048 * 4);
  float*  c1s    = (float*) alloc((size_t)1024 * 2048 * 4);
  bf16_t* hcat   = (bf16_t*)alloc((size_t)DB * (2 * DHE) * 2);
  float*  muRaw  = (float*) alloc((size_t)DB * DZ * 4);
  float*  sigRaw = (float*) alloc((size_t)DB * DZ * 4);
  bf16_t* zb     = (bf16_t*)alloc((size_t)DB * DZ * 2);
  float*  gihC   = (float*) alloc((size_t)DB * 4096 * 4);
  bf16_t* yc0    = (bf16_t*)alloc((size_t)DU * DB * DHC * 2);
  bf16_t* yc1    = (bf16_t*)alloc((size_t)DU * DB * DHC * 2);
  float*  cRaw   = (float*) alloc((size_t)DU * DB * 512 * 4);
  bf16_t* cb     = (bf16_t*)alloc((size_t)DU * DB * 512 * 2);
  float*  linTmp = (float*) alloc((size_t)1024 * 1024 * 4);
  float*  cpart  = (float*) alloc((size_t)1024 * 4096 * 4);
  bf16_t* ys     = (bf16_t*)alloc((size_t)DT * DB * DHD * 2);        // 32 MB
  float*  logits = (float*) alloc((size_t)DT * DB * DV * 4);

  auto cvt = [&](const float* s, bf16_t* d, int rows, int K, int srcStride, int Kpad) {
    long total = (long)rows * Kpad;
    cvt_pad_k<<<dim3((unsigned)((total + 255) / 256)), dim3(256), 0, stream>>>(
        s, d, rows, K, srcStride, Kpad);
  };
  auto gemm = [&](const bf16_t* A, int lda, const bf16_t* W, int ldw,
                  float* C, int ldc, int M, int N, int K,
                  const float* bias, const float* Cin, int mode) {
    dim3 g((unsigned)((N + 255) / 256), (unsigned)(M / 64));
    gemm_bf16_k<<<g, dim3(256), 0, stream>>>(A, lda, W, ldw, Cin, bias, C, ldc,
                                             N, K, mode);
  };
  auto cell = [&](float* gt, float* cst, bf16_t* hst, bf16_t* sec, int secMod,
                  long secBlk, long secRow, int M, int H) {
    lstm_cell_k<<<dim3((unsigned)((M * H + 255) / 256)), dim3(256), 0, stream>>>(
        gt, cst, hst, sec, secMod, secBlk, secRow, M, H);
  };

  // ---- 0) weight conversions ----
  cvt(e0f_Whh, wE0fWhh, 8192, 2048, 2048, 2048);
  cvt(e0b_Whh, wE0bWhh, 8192, 2048, 2048, 2048);
  cvt(e1f_Wih, wE1fWih, 8192, 4096, 4096, 4096);
  cvt(e1f_Whh, wE1fWhh, 8192, 2048, 2048, 2048);
  cvt(e1b_Wih, wE1bWih, 8192, 4096, 4096, 4096);
  cvt(e1b_Whh, wE1bWhh, 8192, 2048, 2048, 2048);
  cvt(mu_W,  wMu,  512, 4096, 4096, 4096);
  cvt(sig_W, wSig, 512, 4096, 4096, 4096);
  cvt(ci_W,  wCi, 1024, 512, 512, 512);
  cvt(c0_Wih, wC0Wih, 4096, 258, 258, 288);
  cvt(c0_Whh, wC0Whh, 4096, 1024, 1024, 1024);
  cvt(c1_Wih, wC1Wih, 4096, 1024, 1024, 1024);
  cvt(c1_Whh, wC1Whh, 4096, 1024, 1024, 1024);
  cvt(co_W, wCo, 512, 1024, 1024, 1024);
  cvt(di_W, wDi, 1024, 512, 512, 512);
  cvt(d0_Wih, wD0Wc, 4096, 512, 770, 512);   // first 512 cols only (c part)
  cvt(d0_Whh, wD0Whh, 4096, 1024, 1024, 1024);
  cvt(d1_Wih, wD1Wih, 4096, 1024, 1024, 1024);
  cvt(d1_Whh, wD1Whh, 4096, 1024, 1024, 1024);
  cvt(head_W, wHead, 258, 1024, 1024, 1024);
  cvt(cond, condB, 512, 258, 258, 288);

  const unsigned encGrid = (unsigned)((DB * 4 * DHE) / 256);

  // ---- 1) encoder layer 0, both directions ----
  for (int dir = 0; dir < 2; ++dir) {
    const float*  Wih = dir ? e0b_Wih : e0f_Wih;
    const float*  bb  = dir ? e0b_b   : e0f_b;
    const bf16_t* Whh = dir ? wE0bWhh : wE0fWhh;
    hipMemsetAsync(h0s, 0, (size_t)DB * DHE * 2, stream);
    hipMemsetAsync(c0s, 0, (size_t)DB * DHE * 4, stream);
    for (int t = 0; t < DT; ++t) {
      int tt = dir ? (DT - 1 - t) : t;   // input column AND y0 slot
      enc_gate_init_k<<<encGrid, 256, 0, stream>>>(gates, Wih, DV, bb, x, DT, tt,
                                                   4 * DHE);
      gemm(h0s, DHE, Whh, DHE, gates, 4 * DHE, DB, 4 * DHE, DHE, nullptr, gates, 1);
      bf16_t* sec = y0 + (size_t)tt * DB * (2 * DHE) + (dir ? DHE : 0);
      cell(gates, c0s, h0s, sec, DB, 0, 2 * DHE, DB, DHE);
    }
  }

  // ---- 2) encoder layer 1, both directions (only final h kept) ----
  for (int dir = 0; dir < 2; ++dir) {
    const bf16_t* Wih = dir ? wE1bWih : wE1fWih;
    const bf16_t* Whh = dir ? wE1bWhh : wE1fWhh;
    const float*  bb  = dir ? e1b_b   : e1f_b;
    hipMemsetAsync(h0s, 0, (size_t)DB * DHE * 2, stream);
    hipMemsetAsync(c0s, 0, (size_t)DB * DHE * 4, stream);
    for (int t = 0; t < DT; ++t) {
      int tt = dir ? (DT - 1 - t) : t;
      const bf16_t* A = y0 + (size_t)tt * DB * (2 * DHE);
      gemm(A, 2 * DHE, Wih, 2 * DHE, gates, 4 * DHE, DB, 4 * DHE, 2 * DHE, bb,
           nullptr, 0);
      gemm(h0s, DHE, Whh, DHE, gates, 4 * DHE, DB, 4 * DHE, DHE, nullptr, gates, 1);
      bf16_t* sec = (t == DT - 1) ? (hcat + (dir ? DHE : 0)) : nullptr;
      cell(gates, c0s, h0s, sec, DB, 0, 2 * DHE, DB, DHE);
    }
  }

  // ---- 3) VAE head: mu, sigma, z ----
  gemm(hcat, 2 * DHE, wMu,  2 * DHE, muRaw,  DZ, DB, DZ, 2 * DHE, mu_b,  nullptr, 0);
  gemm(hcat, 2 * DHE, wSig, 2 * DHE, sigRaw, DZ, DB, DZ, 2 * DHE, sig_b, nullptr, 0);
  vae_head_k<<<(DB * DZ) / 256, 256, 0, stream>>>(muRaw, sigRaw, eps, outMu, outSig,
                                                  zb, (long)DB * DZ);

  // ---- 4) conductor ----
  // h1c = tanh(z @ ci_W^T + ci_b)  -> layer-0 h init; layer-1 zeros
  gemm(zb, DZ, wCi, DZ, linTmp, DHC, DB, DHC, DZ, ci_b, nullptr, 0);
  tanh_bf16_k<<<(DB * DHC) / 256, 256, 0, stream>>>(linTmp, h0s, (long)DB * DHC);
  hipMemsetAsync(c0s, 0, (size_t)DB * DHC * 4, stream);
  hipMemsetAsync(h1s, 0, (size_t)DB * DHC * 2, stream);
  hipMemsetAsync(c1s, 0, (size_t)DB * DHC * 4, stream);
  // constant input gates (broadcast cond_input): gihC = cond @ c0_Wih^T + b
  gemm(condB, 288, wC0Wih, 288, gihC, 4 * DHC, DB, 4 * DHC, 288, c0_b, nullptr, 0);
  for (int u = 0; u < DU; ++u) {
    gemm(h0s, DHC, wC0Whh, DHC, gates, 4 * DHC, DB, 4 * DHC, DHC, nullptr, gihC, 1);
    cell(gates, c0s, h0s, yc0 + (size_t)u * DB * DHC, DB, 0, DHC, DB, DHC);
    gemm(yc0 + (size_t)u * DB * DHC, DHC, wC1Wih, DHC, gates, 4 * DHC, DB, 4 * DHC,
         DHC, c1_b, nullptr, 0);
    gemm(h1s, DHC, wC1Whh, DHC, gates, 4 * DHC, DB, 4 * DHC, DHC, nullptr, gates, 1);
    cell(gates, c1s, h1s, yc1 + (size_t)u * DB * DHC, DB, 0, DHC, DB, DHC);
  }
  // c = yc1 @ co_W^T + co_b  (batched over (u,b): M=1024)
  gemm(yc1, DHC, wCo, DHC, cRaw, 512, DU * DB, 512, DHC, co_b, nullptr, 0);
  cvt(cRaw, cb, DU * DB, 512, 512, 512);

  // ---- 5) decoder (both segments batched: M = U*B = 1024) ----
  gemm(cb, 512, wDi, 512, linTmp, DHD, DU * DB, DHD, 512, di_b, nullptr, 0);
  tanh_bf16_k<<<(DU * DB * DHD) / 256, 256, 0, stream>>>(linTmp, h0s,
                                                         (long)DU * DB * DHD);
  hipMemsetAsync(c0s, 0, (size_t)DU * DB * DHD * 4, stream);
  hipMemsetAsync(h1s, 0, (size_t)DU * DB * DHD * 2, stream);
  hipMemsetAsync(c1s, 0, (size_t)DU * DB * DHD * 4, stream);
  // cpart = c @ d0_Wih[:, :512]^T + d0_b  (constant within each segment)
  gemm(cb, 512, wD0Wc, 512, cpart, 4 * DHD, DU * DB, 4 * DHD, 512, d0_b, nullptr, 0);
  const unsigned decGrid = (unsigned)((DU * DB * 4 * DHD) / 256);
  for (int s = 0; s < DSUB; ++s) {
    dec_gate_init_k<<<decGrid, 256, 0, stream>>>(gates, cpart, d0_Wih, 512 + DV, 512,
                                                 x, DT, s, 4 * DHD);
    gemm(h0s, DHD, wD0Whh, DHD, gates, 4 * DHD, DU * DB, 4 * DHD, DHD, nullptr,
         gates, 1);
    cell(gates, c0s, h0s, nullptr, DU * DB, 0, DHD, DU * DB, DHD);
    gemm(h0s, DHD, wD1Wih, DHD, gates, 4 * DHD, DU * DB, 4 * DHD, DHD, d1_b,
         nullptr, 0);
    gemm(h1s, DHD, wD1Whh, DHD, gates, 4 * DHD, DU * DB, 4 * DHD, DHD, nullptr,
         gates, 1);
    // ys row = t*B + b = u*(16*B) + s*B + b;  m = u*B + b
    cell(gates, c1s, h1s, ys + (size_t)s * DB * DHD, DB, (long)DSUB * DB, DHD,
         DU * DB, DHD);
  }

  // ---- 6) output head + log_softmax (with [t,b] -> [b,t] transpose) ----
  gemm(ys, DHD, wHead, DHD, logits, DV, DT * DB, DV, DHD, head_b, nullptr, 0);
  logsoftmax_k<<<DT * DB, 128, 0, stream>>>(logits, out, DT, DB, DV);
}
